// EncoderDecoder1DBlock_70385924047013
// MI455X (gfx1250) — compile-verified, matrix operand order checked by reading, not run on hardware
//
#include <hip/hip_runtime.h>
#include <hip/hip_bf16.h>

// ---------------------------------------------------------------------------
// Types for CDNA5 WMMA (wave32): 16x16x32 f16 -> f32 accumulate
// ---------------------------------------------------------------------------
typedef __attribute__((ext_vector_type(16))) _Float16 v16h;
typedef __attribute__((ext_vector_type(8)))  _Float16 v8h;
typedef __attribute__((ext_vector_type(8)))  float    v8f;
typedef __attribute__((ext_vector_type(4)))  unsigned int u32x4;
typedef __attribute__((ext_vector_type(4)))  int      i32x4;
typedef __attribute__((ext_vector_type(8)))  int      i32x8;

#if defined(__HIP_DEVICE_COMPILE__) && __has_builtin(__builtin_amdgcn_tensor_load_to_lds)
#define USE_TDM 1
#else
#define USE_TDM 0
#endif

// Compile-time probe (visible in stderr): tells us whether the TDM builtin
// was available on the device pass.
#if defined(__HIP_DEVICE_COMPILE__) && !USE_TDM
#warning "CDNA5 probe: __builtin_amdgcn_tensor_load_to_lds NOT available - using manual LDS staging"
#endif

__device__ __forceinline__ v8f wmma_f32_16x16x32(v16h a, v16h b, v8f c) {
    // (neg_a, A, neg_b, B, c_mod, C, reuse_a, reuse_b)
    return __builtin_amdgcn_wmma_f32_16x16x32_f16(false, a, false, b,
                                                  (short)0, c, false, false);
}

// Load a 16-half fragment as two contiguous 16B chunks (global_load_b128 path).
// CDNA5 16-bit A-fragment per lane: halves 0..7 = K[koff..koff+7],
// halves 8..15 = K[koff+16..koff+23]  (koff = 8*(lane>=16)); B mirrors A.
__device__ __forceinline__ v16h load_frag(const _Float16* p0, const _Float16* p1) {
    v8h lo = *(const v8h*)p0;
    v8h hi = *(const v8h*)p1;
    v16h r;
#pragma unroll
    for (int i = 0; i < 8; ++i) { r[i] = lo[i]; r[8 + i] = hi[i]; }
    return r;
}

#if USE_TDM
// Raw LDS byte offset of a __shared__ object (addrspace(3) pointers are the
// 32-bit LDS offsets on AMDGPU).
__device__ __forceinline__ unsigned lds_offset_of(const void* p) {
    return (unsigned)(unsigned long long)
        (__attribute__((address_space(3))) const void*)p;
}

// TDM: async-load a 2D tile of 32 rows x 64 halves (row stride = pitch
// elements) from global to LDS at lds_off, padding each 128B LDS row by 32B
// (-> 80-half LDS pitch). Tracked by TENSORcnt.
__device__ __forceinline__ void tdm_load_v_tile(unsigned lds_off,
                                                const _Float16* gsrc,
                                                int pitch) {
    unsigned long long ga = (unsigned long long)gsrc;
    u32x4 g0;
    g0[0] = 1u;                                      // count=1 (valid user D#)
    g0[1] = lds_off;                                 // lds_addr
    g0[2] = (unsigned)(ga & 0xffffffffu);            // global_addr[31:0]
    g0[3] = (unsigned)((ga >> 32) & 0x1ffffffu)      // global_addr[56:32]
          | (2u << 30);                              // type=2 ("image")
    i32x8 g1;
    g1[0] = (1 << 16)                                // data_size = 2 bytes
          | (1 << 20)                                // pad_enable
          | (4 << 22)                                // pad_interval: 32 dwords
          | (7 << 25);                               // pad_amount: 8 dwords
    g1[1] = (64 << 16);                              // tensor_dim0 = 64
    g1[2] = (32 << 16);                              // tensor_dim1 = 32
    g1[3] = (64 << 16);                              // tile_dim0 = 64
    g1[4] = 32;                                      // tile_dim1 = 32
    g1[5] = pitch;                                   // tensor_dim0_stride lo
    g1[6] = 0;
    g1[7] = 0;
    i32x4 z4 = {};
#if defined(__clang_major__) && (__clang_major__ >= 23)
    i32x8 z8 = {};
    __builtin_amdgcn_tensor_load_to_lds(g0, g1, z4, z4, z8, 0);
#else
    __builtin_amdgcn_tensor_load_to_lds(g0, g1, z4, z4, 0);
#endif
}
#endif  // USE_TDM

// ---------------------------------------------------------------------------
// Weight conversion: f32 [K,N] -> f16 [N,K] (transposed so B-fragments are
// contiguous along K)
// ---------------------------------------------------------------------------
__global__ void transpose_to_f16(const float* __restrict__ w,
                                 _Float16* __restrict__ wt, int K, int N) {
    int idx = blockIdx.x * blockDim.x + threadIdx.x;
    int total = K * N;
    if (idx >= total) return;
    int n = idx / K;
    int k = idx - n * K;
    wt[(size_t)n * K + k] = (_Float16)w[(size_t)k * N + n];
}

__global__ void to_f16(const float* __restrict__ in,
                       _Float16* __restrict__ out, int n) {
    int idx = blockIdx.x * blockDim.x + threadIdx.x;
    if (idx < n) out[idx] = (_Float16)in[idx];
}

// ---------------------------------------------------------------------------
// LayerNorm: one block per row, writes f16 (A-operand for following GEMM)
// ---------------------------------------------------------------------------
__global__ void layernorm_f16(const float* __restrict__ x,
                              const float* __restrict__ scale,
                              const float* __restrict__ bias,
                              _Float16* __restrict__ out, int D) {
    const int row = blockIdx.x;
    const float* xr = x + (size_t)row * D;
    __shared__ float red[64];
    float s = 0.f, s2 = 0.f;
    for (int i = threadIdx.x; i < D; i += blockDim.x) {
        float v = xr[i];
        s += v; s2 += v * v;
    }
#pragma unroll
    for (int off = 16; off > 0; off >>= 1) {
        s  += __shfl_xor(s,  off, 32);
        s2 += __shfl_xor(s2, off, 32);
    }
    int wid = threadIdx.x >> 5;
    int nw  = blockDim.x >> 5;
    if ((threadIdx.x & 31) == 0) { red[wid] = s; red[wid + 32] = s2; }
    __syncthreads();
    if (threadIdx.x == 0) {
        float ts = 0.f, ts2 = 0.f;
        for (int w = 0; w < nw; ++w) { ts += red[w]; ts2 += red[w + 32]; }
        red[0] = ts; red[32] = ts2;
    }
    __syncthreads();
    float mean = red[0] / (float)D;
    float var  = red[32] / (float)D - mean * mean;
    float rstd = rsqrtf(var + 1e-6f);
    for (int i = threadIdx.x; i < D; i += blockDim.x) {
        float v = (xr[i] - mean) * rstd * scale[i] + bias[i];
        out[(size_t)row * D + i] = (_Float16)v;
    }
}

// ---------------------------------------------------------------------------
// Generic WMMA GEMM: C[M,N] = relu?(alpha * A[M,K] @ Bt[N,K]^T + bias) + res
// A, Bt in f16. One wave computes a 32x64 tile: 2x4 grid of 16x16 WMMA
// accumulators. All 6 fragment loads issued before the 8 WMMAs per K-step so
// the scheduler can pipeline waits. 8 waves per 256-thread block.
// ---------------------------------------------------------------------------
__global__ void gemm_f16_wmma(const _Float16* __restrict__ A,
                              const _Float16* __restrict__ Bt,
                              const float* __restrict__ bias,
                              const float* __restrict__ residual,
                              float* __restrict__ outF,
                              _Float16* __restrict__ outH,
                              int M, int N, int K, float alpha, int relu) {
    const int lane = threadIdx.x & 31;
    const int wave = threadIdx.x >> 5;
    const int tiles_n = N >> 6;                 // 64-wide tiles
    const int tiles   = (M >> 5) * tiles_n;     // 32-tall tiles
    int tile = blockIdx.x * (blockDim.x >> 5) + wave;
    if (tile >= tiles) return;                  // uniform per-wave exit
    const int mt = tile / tiles_n;
    const int nt = tile - mt * tiles_n;

    const int half_id = lane >> 4;
    const int l16     = lane & 15;
    const int koff    = half_id * 8;

    const _Float16* arow0 = A + (size_t)(mt * 32 + l16) * K + koff;
    const _Float16* arow1 = arow0 + (size_t)16 * K;
    const _Float16* brow  = Bt + (size_t)(nt * 64 + l16) * K + koff;

    v8f acc[2][4] = {};
#pragma unroll 2
    for (int kk = 0; kk < K; kk += 32) {
        __builtin_prefetch(arow0 + kk + 256, 0, 0);  // global_prefetch_b8
        v16h a0 = load_frag(arow0 + kk, arow0 + kk + 16);
        v16h a1 = load_frag(arow1 + kk, arow1 + kk + 16);
        v16h bf[4];
#pragma unroll
        for (int j = 0; j < 4; ++j) {
            const _Float16* bp = brow + (size_t)j * 16 * K + kk;
            bf[j] = load_frag(bp, bp + 16);
        }
#pragma unroll
        for (int j = 0; j < 4; ++j) {
            acc[0][j] = wmma_f32_16x16x32(a0, bf[j], acc[0][j]);
            acc[1][j] = wmma_f32_16x16x32(a1, bf[j], acc[1][j]);
        }
    }

#pragma unroll
    for (int i = 0; i < 2; ++i) {
#pragma unroll
        for (int j = 0; j < 4; ++j) {
            const int col = nt * 64 + j * 16 + l16;
            const float bc = bias ? bias[col] : 0.f;
#pragma unroll
            for (int r = 0; r < 8; ++r) {
                int row = mt * 32 + i * 16 + r + half_id * 8;
                float f = acc[i][j][r] * alpha + bc;
                if (relu) f = fmaxf(f, 0.f);
                size_t o = (size_t)row * N + col;
                if (residual) f += residual[o];
                if (outF) outF[o] = f;
                if (outH) outH[o] = (_Float16)f;
            }
        }
    }
}

// ---------------------------------------------------------------------------
// Flash attention: one wave per (batch, head, 16-row q tile).
// Q pre-scaled by 1/sqrt(Dh) in the projection GEMM.
// Online softmax in C-register layout; P reshaped through LDS to A layout;
// V tile moved to LDS by the Tensor Data Mover (TENSORcnt) when available,
// else by coalesced b128 stores; fragments gathered from LDS.
// ---------------------------------------------------------------------------
__global__ void flash_attn_wmma(const _Float16* __restrict__ Q,
                                const _Float16* __restrict__ Kt,
                                const _Float16* __restrict__ V,
                                _Float16* __restrict__ O,
                                int S, int Sk, int D, int causal) {
    const int qt = blockIdx.x, h = blockIdx.y, b = blockIdx.z;
    const int lane    = threadIdx.x;
    const int half_id = lane >> 4;
    const int l16     = lane & 15;
    const int koff    = half_id * 8;
    const int q0      = qt * 16;

    const _Float16* qrow = Q + (size_t)b * S * D + (size_t)(q0 + l16) * D + h * 64 + koff;
    v16h qa0 = load_frag(qrow,      qrow + 16);   // dh 0..31
    v16h qa1 = load_frag(qrow + 32, qrow + 48);   // dh 32..63

    float mrun[8], lrun[8];
    v8f oacc[4] = {};
#pragma unroll
    for (int r = 0; r < 8; ++r) { mrun[r] = -3.0e38f; lrun[r] = 0.f; }

    __shared__ _Float16 pls[16][32];      // P tile (C-layout -> A-layout bounce)
    __shared__ _Float16 vls[32][80];      // V tile 32 keys x 64 dh, 160B row pitch

    const int kt_end = causal ? (q0 + 47) / 32 : (Sk / 32);
    const _Float16* kbatch = Kt + (size_t)b * Sk * D + h * 64;
    const _Float16* vbatch = V  + (size_t)b * Sk * D + h * 64;

#if USE_TDM
    const unsigned vls_off = lds_offset_of(&vls[0][0]);
#endif

    for (int kt = 0; kt < kt_end; ++kt) {
        const int kbase = kt * 32;

        // --- stage V tile to LDS ---
#if USE_TDM
        // Ensure previous iteration's LDS gathers are done before the TDM
        // overwrites the tile, then launch the async tensor DMA.
        asm volatile("s_wait_dscnt 0x0" ::: "memory");
        tdm_load_v_tile(vls_off, vbatch + (size_t)kbase * D, D);
#else
        __syncthreads();
#pragma unroll
        for (int it = 0; it < 8; ++it) {
            int c      = it * 32 + lane;     // 256 chunks of 8 halves
            int vrow   = c >> 3;             // 8 chunks per 64-half row
            int vcoff  = (c & 7) * 8;
            *(v8h*)&vls[vrow][vcoff] =
                *(const v8h*)(vbatch + (size_t)(kbase + vrow) * D + vcoff);
        }
#endif

        // --- scores: S = Q (16x64) x K^T (64x32), two 16-col subtiles ---
        const _Float16* krow0 = kbatch + (size_t)(kbase + l16) * D + koff;
        const _Float16* krow1 = krow0 + (size_t)16 * D;
        v16h kb00 = load_frag(krow0,      krow0 + 16);
        v16h kb01 = load_frag(krow0 + 32, krow0 + 48);
        v16h kb10 = load_frag(krow1,      krow1 + 16);
        v16h kb11 = load_frag(krow1 + 32, krow1 + 48);
        v8f s0 = {}, s1 = {};
        s0 = wmma_f32_16x16x32(qa0, kb00, s0);
        s0 = wmma_f32_16x16x32(qa1, kb01, s0);
        s1 = wmma_f32_16x16x32(qa0, kb10, s1);
        s1 = wmma_f32_16x16x32(qa1, kb11, s1);

        // --- online softmax (row M = r + 8*half_id, col N = lane&15) ---
        float p0[8], p1[8];
#pragma unroll
        for (int r = 0; r < 8; ++r) {
            const int rowq = q0 + r + half_id * 8;
            float a = s0[r], cc = s1[r];
            if (causal) {
                if (kbase + l16      > rowq) a  = -1.0e9f;
                if (kbase + 16 + l16 > rowq) cc = -1.0e9f;
            }
            float tm = fmaxf(a, cc);
#pragma unroll
            for (int off = 8; off > 0; off >>= 1)
                tm = fmaxf(tm, __shfl_xor(tm, off, 32));   // within 16-lane half
            const float mnew = fmaxf(mrun[r], tm);
            const float resc = __expf(mrun[r] - mnew);
            mrun[r] = mnew;
            const float e0 = __expf(a - mnew);
            const float e1 = __expf(cc - mnew);
            float ls = e0 + e1;
#pragma unroll
            for (int off = 8; off > 0; off >>= 1)
                ls += __shfl_xor(ls, off, 32);
            lrun[r] = lrun[r] * resc + ls;
#pragma unroll
            for (int d = 0; d < 4; ++d) oacc[d][r] *= resc;
            p0[r] = e0; p1[r] = e1;
        }

        // --- C-layout -> A-layout reshape of P through LDS ---
#pragma unroll
        for (int r = 0; r < 8; ++r) {
            pls[r + half_id * 8][l16]      = (_Float16)p0[r];
            pls[r + half_id * 8][16 + l16] = (_Float16)p1[r];
        }
        __syncthreads();
        v16h pa = load_frag(&pls[l16][koff], &pls[l16][koff + 16]);

#if USE_TDM
        __builtin_amdgcn_s_wait_tensorcnt(0);   // V tile resident in LDS
#endif

        // --- O += P (16x32) x V (32x64), V fragments gathered from LDS ---
#pragma unroll
        for (int d = 0; d < 4; ++d) {
            v16h vb;
#pragma unroll
            for (int i = 0; i < 8; ++i) {
                vb[i]     = vls[koff + i][d * 16 + l16];
                vb[8 + i] = vls[koff + 16 + i][d * 16 + l16];
            }
            oacc[d] = wmma_f32_16x16x32(pa, vb, oacc[d]);
        }
    }

    // --- epilogue: O /= l ---
#pragma unroll
    for (int r = 0; r < 8; ++r) {
        const float inv  = 1.f / lrun[r];
        const int   rowq = q0 + r + half_id * 8;
#pragma unroll
        for (int d = 0; d < 4; ++d) {
            O[(size_t)b * S * D + (size_t)rowq * D + h * 64 + d * 16 + l16] =
                (_Float16)(oacc[d][r] * inv);
        }
    }
}

// ---------------------------------------------------------------------------
// Host orchestration
// ---------------------------------------------------------------------------
extern "C" void kernel_launch(void* const* d_in, const int* in_sizes, int n_in,
                              void* d_out, int out_size, void* d_ws, size_t ws_size,
                              hipStream_t stream) {
    (void)in_sizes; (void)n_in; (void)out_size; (void)ws_size;
    constexpr int B = 2, S = 2048, D = 1024, F = 4096, H = 16;
    constexpr int M = B * S;
    const float qscale = 0.125f;  // 1/sqrt(Dh=64)

    const float* targets = (const float*)d_in[0];
    const float* encoded = (const float*)d_in[1];
    const float* ln1_s = (const float*)d_in[2];
    const float* ln1_b = (const float*)d_in[3];
    const float* sa_wq = (const float*)d_in[4];
    const float* sa_wk = (const float*)d_in[5];
    const float* sa_wv = (const float*)d_in[6];
    const float* sa_wo = (const float*)d_in[7];
    const float* ln2_s = (const float*)d_in[8];
    const float* ln2_b = (const float*)d_in[9];
    const float* ca_wq = (const float*)d_in[10];
    const float* ca_wk = (const float*)d_in[11];
    const float* ca_wv = (const float*)d_in[12];
    const float* ca_wo = (const float*)d_in[13];
    const float* ln3_s = (const float*)d_in[14];
    const float* ln3_b = (const float*)d_in[15];
    const float* mlp_w1 = (const float*)d_in[16];
    const float* mlp_b1 = (const float*)d_in[17];
    const float* mlp_w2 = (const float*)d_in[18];
    const float* mlp_b2 = (const float*)d_in[19];
    float* out = (float*)d_out;

    // ---- workspace carve ----
    char* p = (char*)d_ws;
    auto alloc = [&](size_t bytes) -> void* {
        void* r = (void*)p;
        p += (bytes + 255) & ~(size_t)255;
        return r;
    };
    _Float16* sa_wq_t = (_Float16*)alloc((size_t)D * D * 2);
    _Float16* sa_wk_t = (_Float16*)alloc((size_t)D * D * 2);
    _Float16* sa_wv_t = (_Float16*)alloc((size_t)D * D * 2);
    _Float16* sa_wo_t = (_Float16*)alloc((size_t)D * D * 2);
    _Float16* ca_wq_t = (_Float16*)alloc((size_t)D * D * 2);
    _Float16* ca_wk_t = (_Float16*)alloc((size_t)D * D * 2);
    _Float16* ca_wv_t = (_Float16*)alloc((size_t)D * D * 2);
    _Float16* ca_wo_t = (_Float16*)alloc((size_t)D * D * 2);
    _Float16* w1_t    = (_Float16*)alloc((size_t)D * F * 2);
    _Float16* w2_t    = (_Float16*)alloc((size_t)F * D * 2);
    _Float16* xln     = (_Float16*)alloc((size_t)M * D * 2);
    _Float16* enc_h   = (_Float16*)alloc((size_t)M * D * 2);
    _Float16* qb      = (_Float16*)alloc((size_t)M * D * 2);
    _Float16* kb      = (_Float16*)alloc((size_t)M * D * 2);
    _Float16* vb      = (_Float16*)alloc((size_t)M * D * 2);
    _Float16* attn    = (_Float16*)alloc((size_t)M * D * 2);
    float*    xbuf    = (float*)alloc((size_t)M * D * 4);
    float*    ybuf    = (float*)alloc((size_t)M * D * 4);
    _Float16* hbuf    = (_Float16*)alloc((size_t)M * F * 2);

    auto cdiv = [](int a, int b) { return (a + b - 1) / b; };
    // 32x64 tiles per wave, 8 waves per block
    auto ggrid = [&](int m, int n) { return cdiv((m / 32) * (n / 64), 8); };

    // ---- weight conversion (f32 [K,N] -> f16 [N,K]) ----
    transpose_to_f16<<<cdiv(D * D, 256), 256, 0, stream>>>(sa_wq, sa_wq_t, D, D);
    transpose_to_f16<<<cdiv(D * D, 256), 256, 0, stream>>>(sa_wk, sa_wk_t, D, D);
    transpose_to_f16<<<cdiv(D * D, 256), 256, 0, stream>>>(sa_wv, sa_wv_t, D, D);
    transpose_to_f16<<<cdiv(D * D, 256), 256, 0, stream>>>(sa_wo, sa_wo_t, D, D);
    transpose_to_f16<<<cdiv(D * D, 256), 256, 0, stream>>>(ca_wq, ca_wq_t, D, D);
    transpose_to_f16<<<cdiv(D * D, 256), 256, 0, stream>>>(ca_wk, ca_wk_t, D, D);
    transpose_to_f16<<<cdiv(D * D, 256), 256, 0, stream>>>(ca_wv, ca_wv_t, D, D);
    transpose_to_f16<<<cdiv(D * D, 256), 256, 0, stream>>>(ca_wo, ca_wo_t, D, D);
    transpose_to_f16<<<cdiv(D * F, 256), 256, 0, stream>>>(mlp_w1, w1_t, D, F);
    transpose_to_f16<<<cdiv(F * D, 256), 256, 0, stream>>>(mlp_w2, w2_t, F, D);
    to_f16<<<cdiv(M * D, 256), 256, 0, stream>>>(encoded, enc_h, M * D);

    // ---- self-attention (causal) ----
    layernorm_f16<<<M, 256, 0, stream>>>(targets, ln1_s, ln1_b, xln, D);
    gemm_f16_wmma<<<ggrid(M, D), 256, 0, stream>>>(xln, sa_wq_t, nullptr, nullptr,
                                                   nullptr, qb, M, D, D, qscale, 0);
    gemm_f16_wmma<<<ggrid(M, D), 256, 0, stream>>>(xln, sa_wk_t, nullptr, nullptr,
                                                   nullptr, kb, M, D, D, 1.f, 0);
    gemm_f16_wmma<<<ggrid(M, D), 256, 0, stream>>>(xln, sa_wv_t, nullptr, nullptr,
                                                   nullptr, vb, M, D, D, 1.f, 0);
    flash_attn_wmma<<<dim3(S / 16, H, B), 32, 0, stream>>>(qb, kb, vb, attn,
                                                           S, S, D, /*causal=*/1);
    gemm_f16_wmma<<<ggrid(M, D), 256, 0, stream>>>(attn, sa_wo_t, nullptr, targets,
                                                   xbuf, nullptr, M, D, D, 1.f, 0);

    // ---- cross-attention over encoder output ----
    layernorm_f16<<<M, 256, 0, stream>>>(xbuf, ln2_s, ln2_b, xln, D);
    gemm_f16_wmma<<<ggrid(M, D), 256, 0, stream>>>(xln, ca_wq_t, nullptr, nullptr,
                                                   nullptr, qb, M, D, D, qscale, 0);
    gemm_f16_wmma<<<ggrid(M, D), 256, 0, stream>>>(enc_h, ca_wk_t, nullptr, nullptr,
                                                   nullptr, kb, M, D, D, 1.f, 0);
    gemm_f16_wmma<<<ggrid(M, D), 256, 0, stream>>>(enc_h, ca_wv_t, nullptr, nullptr,
                                                   nullptr, vb, M, D, D, 1.f, 0);
    flash_attn_wmma<<<dim3(S / 16, H, B), 32, 0, stream>>>(qb, kb, vb, attn,
                                                           S, S, D, /*causal=*/0);
    gemm_f16_wmma<<<ggrid(M, D), 256, 0, stream>>>(attn, ca_wo_t, nullptr, xbuf,
                                                   ybuf, nullptr, M, D, D, 1.f, 0);

    // ---- MLP ----
    layernorm_f16<<<M, 256, 0, stream>>>(ybuf, ln3_s, ln3_b, xln, D);
    gemm_f16_wmma<<<ggrid(M, F), 256, 0, stream>>>(xln, w1_t, mlp_b1, nullptr,
                                                   nullptr, hbuf, M, F, D, 1.f, 1);
    gemm_f16_wmma<<<ggrid(M, D), 256, 0, stream>>>(hbuf, w2_t, mlp_b2, ybuf,
                                                   out, nullptr, M, D, F, 1.f, 0);
}
